// DynamicConv2d_50062138802577
// MI455X (gfx1250) — matrix operand order repaired and checked by reading, not compiled, not run on hardware
//
#include <hip/hip_runtime.h>
#include <hip/hip_bf16.h>
#include <stdint.h>

typedef __attribute__((ext_vector_type(16))) _Float16 v16h;
typedef __attribute__((ext_vector_type(8)))  float    v8f;
typedef __attribute__((ext_vector_type(4)))  int      v4i;

#define NB    32
#define NC    128
#define NW    56
#define NHW   3136      // 56*56
#define NOUT  256
#define KRED  1152      // 9 taps * 128 channels, k = tap*128 + c
#define NCHUNK 36       // 1152 / 32
#define ALDS  40        // padded row stride in halves (80B -> bank-spread)

// ---- CDNA5 async global->LDS copy (16B per lane), with safe fallback ----
__device__ __forceinline__ void async_copy16(const _Float16* g, _Float16* l) {
#if defined(__AMDGCN__) && __has_builtin(__builtin_amdgcn_global_load_async_to_lds_b128)
    __builtin_amdgcn_global_load_async_to_lds_b128(
        (__attribute__((address_space(1))) v4i*)(g),
        (__attribute__((address_space(3))) v4i*)(l), 0, 0);
#else
    *(uint4*)l = *(const uint4*)g;
#endif
}

__device__ __forceinline__ void wait_async0() {
#if defined(__AMDGCN__)
#if __has_builtin(__builtin_amdgcn_s_wait_asynccnt)
    __builtin_amdgcn_s_wait_asynccnt(0);
#else
    asm volatile("s_wait_asynccnt 0" ::: "memory");
#endif
#endif
}

// ---------------- Phase 1a: global average pool ----------------
__global__ void pool_kernel(const float* __restrict__ x, float* __restrict__ pooled) {
    __shared__ float red[128];
    const int bc = blockIdx.x;                 // b*128 + c
    const float* p = x + (size_t)bc * NHW;
    float s = 0.f;
    for (int i = threadIdx.x; i < NHW; i += 128) s += p[i];
    red[threadIdx.x] = s;
    __syncthreads();
    for (int off = 64; off > 0; off >>= 1) {
        if ((int)threadIdx.x < off) red[threadIdx.x] += red[threadIdx.x + off];
        __syncthreads();
    }
    if (threadIdx.x == 0) pooled[bc] = red[0] * (1.0f / (float)NHW);
}

// ---------------- Phase 1b: fc1->relu->fc2->softmax ----------------
__global__ void attn_kernel(const float* __restrict__ pooled,
                            const float* __restrict__ fc1_w, const float* __restrict__ fc1_b,
                            const float* __restrict__ fc2_w, const float* __restrict__ fc2_b,
                            float* __restrict__ attn) {
    __shared__ float hdn[64];
    __shared__ float lg[4];
    const int b = blockIdx.x, t = threadIdx.x;
    const float* pv = pooled + b * NC;
    const float* w1 = fc1_w + t * NC;
    float s = fc1_b[t];
    for (int c = 0; c < NC; ++c) s += pv[c] * w1[c];
    hdn[t] = fmaxf(s, 0.f);
    __syncthreads();
    if (t < 4) {
        const float* w2 = fc2_w + t * 64;
        float l = fc2_b[t];
        for (int j = 0; j < 64; ++j) l += hdn[j] * w2[j];
        lg[t] = l;
    }
    __syncthreads();
    if (t == 0) {
        float m = fmaxf(fmaxf(lg[0], lg[1]), fmaxf(lg[2], lg[3]));
        float e0 = __expf(lg[0] - m), e1 = __expf(lg[1] - m);
        float e2 = __expf(lg[2] - m), e3 = __expf(lg[3] - m);
        float inv = 1.0f / (e0 + e1 + e2 + e3);
        attn[b * 4 + 0] = e0 * inv;
        attn[b * 4 + 1] = e1 * inv;
        attn[b * 4 + 2] = e2 * inv;
        attn[b * 4 + 3] = e3 * inv;
    }
}

// ---------------- Phase 2: per-sample weight synthesis -> f16, GEMM layout ----------------
// wsynth[b][o][k] with k = tap*128 + c  (bank layout is [kk][o][c][tap], tap innermost)
__global__ void synth_kernel(const float* __restrict__ bank, const float* __restrict__ attn,
                             _Float16* __restrict__ wsynth) {
    const int idx = blockIdx.x * 256 + threadIdx.x;      // < 32*256*1152, exact
    const int b   = idx / (NOUT * KRED);
    const int rem = idx - b * (NOUT * KRED);
    const int o   = rem / KRED;
    const int k   = rem - o * KRED;
    const int tap = k >> 7;
    const int c   = k & 127;
    const size_t kstride = (size_t)NOUT * NC * 9;
    const size_t base = ((size_t)o * NC + c) * 9 + tap;
    const float a0 = attn[b * 4 + 0], a1 = attn[b * 4 + 1];
    const float a2 = attn[b * 4 + 2], a3 = attn[b * 4 + 3];
    float v = a0 * bank[base]
            + a1 * bank[base + kstride]
            + a2 * bank[base + 2 * kstride]
            + a3 * bank[base + 3 * kstride];
    wsynth[idx] = (_Float16)v;
}

// ---------------- Phase 3: implicit-GEMM conv via v_wmma_f32_16x16x32_f16 ----------------
// Per WG: sample b, M=256 (all out ch) x N=64 (spatial). 8 waves (4x2), each 64x32 output.
// Double-buffered LDS; A tile streamed with GLOBAL_LOAD_ASYNC_TO_LDS_B128 (ASYNCcnt).
__global__ __launch_bounds__(256)
void conv_wmma_kernel(const float* __restrict__ x, const _Float16* __restrict__ wsynth,
                      float* __restrict__ out) {
    __shared__ _Float16 At[2][NOUT * ALDS];  // 2 x 20480 B
    __shared__ _Float16 Bt[2][64 * ALDS];    // 2 x  5120 B

    const int t    = threadIdx.x;
    const int n0   = blockIdx.x * 64;        // spatial tile base (3136 = 49*64)
    const int b    = blockIdx.y;
    const int lane = t & 31, wid = t >> 5;
    const int mi   = (wid & 3) * 64;         // wave M offset
    const int ni   = (wid >> 2) * 32;        // wave N offset
    const int half = lane >> 4, r = lane & 15;

    // per-thread invariant fill indexing
    const int rowA = t >> 2;                 // A: rows rowA + 64*s
    const int segA = (t & 3) * 8;            // halves
    const int nB   = t & 63;                 // B: spatial column (invariant!)
    const int kB   = t >> 6;                 // B: channels kB + 4*s
    const int pos  = n0 + nB;
    const int y0   = pos / NW, x0 = pos % NW;

    v8f acc[4][2] = {};
    const float*    xb = x + (size_t)b * NC * NHW;
    const _Float16* wb = wsynth + (size_t)b * NOUT * KRED;

    // ---- prologue: fill chunk 0 into buffer 0 ----
    {
        #pragma unroll
        for (int s = 0; s < 4; ++s) {
            const int row = rowA + (s << 6);
            async_copy16(wb + (size_t)row * KRED + segA, &At[0][row * ALDS + segA]);
        }
        const int yy = y0 - 1, xx = x0 - 1;                      // tap 0 = (-1,-1)
        const bool ok = (unsigned)yy < (unsigned)NW && (unsigned)xx < (unsigned)NW;
        const float* src = xb + (size_t)kB * NHW + yy * NW + xx;
        _Float16* dst = &Bt[0][nB * ALDS + kB];
        #pragma unroll
        for (int s = 0; s < 8; ++s)
            dst[4 * s] = (_Float16)(ok ? src[(size_t)(4 * s) * NHW] : 0.f);
        wait_async0();
        __syncthreads();
    }

    for (int q = 0; q < NCHUNK; ++q) {
        const int cur = q & 1;

        // ---- prefetch chunk q+1 into the other buffer (async A + VALU im2col B) ----
        if (q + 1 < NCHUNK) {
            const int nxt  = cur ^ 1;
            const int qq   = q + 1;
            const int tap  = qq >> 2;
            const int di   = tap / 3 - 1;
            const int dj   = tap % 3 - 1;
            const int cblk = (qq & 3) << 5;
            #pragma unroll
            for (int s = 0; s < 4; ++s) {
                const int row = rowA + (s << 6);
                async_copy16(wb + (size_t)row * KRED + qq * 32 + segA,
                             &At[nxt][row * ALDS + segA]);
            }
            const int yy = y0 + di, xx = x0 + dj;
            const bool ok = (unsigned)yy < (unsigned)NW && (unsigned)xx < (unsigned)NW;
            const float* src = xb + (size_t)(cblk + kB) * NHW + yy * NW + xx;
            _Float16* dst = &Bt[nxt][nB * ALDS + kB];
            #pragma unroll
            for (int s = 0; s < 8; ++s)
                dst[4 * s] = (_Float16)(ok ? src[(size_t)(4 * s) * NHW] : 0.f);
        }

        // ---- fragment gathers per ISA 7.12.2 layouts (all 16B LDS loads) ----
        union Fr { v16h v; uint4 q2[2]; };
        Fr A[4], B0, B1;
        #pragma unroll
        for (int a = 0; a < 4; ++a) {
            const _Float16* ap = &At[cur][(mi + 16 * a + r) * ALDS + 8 * half];
            A[a].q2[0] = *(const uint4*)(ap);
            A[a].q2[1] = *(const uint4*)(ap + 16);
        }
        {
            const _Float16* b0p = &Bt[cur][(ni + r) * ALDS + 8 * half];
            B0.q2[0] = *(const uint4*)(b0p);
            B0.q2[1] = *(const uint4*)(b0p + 16);
            const _Float16* b1p = &Bt[cur][(ni + 16 + r) * ALDS + 8 * half];
            B1.q2[0] = *(const uint4*)(b1p);
            B1.q2[1] = *(const uint4*)(b1p + 16);
        }
        #pragma unroll
        for (int a = 0; a < 4; ++a) {
            acc[a][0] = __builtin_amdgcn_wmma_f32_16x16x32_f16(false, A[a].v, false, B0.v,
                                                               (short)0, acc[a][0], false, false);
            acc[a][1] = __builtin_amdgcn_wmma_f32_16x16x32_f16(false, A[a].v, false, B1.v,
                                                               (short)0, acc[a][1], false, false);
        }

        wait_async0();      // next A tile landed in LDS
        __syncthreads();    // publish next buffers / retire reads of current
    }

    // --- writeback: D layout, VGPR j -> M = j + 8*half, N = lane%16 ---
    float* ob = out + (size_t)b * NOUT * NHW;
    #pragma unroll
    for (int a = 0; a < 4; ++a)
        #pragma unroll
        for (int bn = 0; bn < 2; ++bn) {
            const int col = n0 + ni + 16 * bn + r;
            #pragma unroll
            for (int j = 0; j < 8; ++j) {
                const int row = mi + 16 * a + 8 * half + j;
                ob[(size_t)row * NHW + col] = acc[a][bn][j];
            }
        }
}

extern "C" void kernel_launch(void* const* d_in, const int* in_sizes, int n_in,
                              void* d_out, int out_size, void* d_ws, size_t ws_size,
                              hipStream_t stream) {
    const float* x     = (const float*)d_in[0];
    const float* bank  = (const float*)d_in[1];
    const float* fc1_w = (const float*)d_in[2];
    const float* fc1_b = (const float*)d_in[3];
    const float* fc2_w = (const float*)d_in[4];
    const float* fc2_b = (const float*)d_in[5];
    float* out = (float*)d_out;

    char* ws = (char*)d_ws;
    float*    pooled = (float*)ws;                 // 32*128*4   = 16 KB
    float*    attn   = (float*)(ws + 16384);       // 32*4*4     = 512 B
    _Float16* wsynth = (_Float16*)(ws + 32768);    // 32*256*1152*2 ≈ 18.9 MB

    pool_kernel <<<NB * NC, 128, 0, stream>>>(x, pooled);
    attn_kernel <<<NB, 64, 0, stream>>>(pooled, fc1_w, fc1_b, fc2_w, fc2_b, attn);
    synth_kernel<<<(NB * NOUT * KRED) / 256, 256, 0, stream>>>(bank, attn, wsynth);
    conv_wmma_kernel<<<dim3(NHW / 64, NB), 256, 0, stream>>>(x, wsynth, out);
}
